// DeepseekMoE_86500641341794
// MI455X (gfx1250) — matrix operand order repaired and checked by reading, not compiled, not run on hardware
//
#include <hip/hip_runtime.h>
#include <cstdint>
#include <cstddef>

#define T_TOK 1024
#define HDIM  2048
#define NEXP  64
#define TOPK  6
#define IDIM  1408
#define ISH   2816
#define CAP   256
#define TKTOT (T_TOK*TOPK)

typedef __attribute__((ext_vector_type(16))) __bf16       v16bf;
typedef __attribute__((ext_vector_type(8)))  float        v8f;
typedef __attribute__((ext_vector_type(4)))  unsigned int u32x4;

union FragB {
    v16bf        v;
    unsigned int ui[8];
    u32x4        q[2];
};

__device__ __forceinline__ unsigned int packbf_rne(float a, float b) {
    unsigned int ua = __float_as_uint(a);
    ua += 0x7FFFu + ((ua >> 16) & 1u);
    unsigned int ub = __float_as_uint(b);
    ub += 0x7FFFu + ((ub >> 16) & 1u);
    return (ua >> 16) | (ub & 0xFFFF0000u);
}

__device__ __forceinline__ unsigned int packbf_fast(float a, float b) {
    unsigned int ua = __float_as_uint(a) + 0x8000u;
    unsigned int ub = __float_as_uint(b) + 0x8000u;
    return (ua >> 16) | (ub & 0xFFFF0000u);
}

__device__ __forceinline__ unsigned short f2bf(float a) {
    unsigned int ua = __float_as_uint(a);
    ua += 0x7FFFu + ((ua >> 16) & 1u);
    return (unsigned short)(ua >> 16);
}

__device__ __forceinline__ float silu_f(float g) {
    return g / (1.0f + __expf(-g));
}

__device__ __forceinline__ v8f wmma_bf16(const FragB& A, const FragB& B, v8f C) {
    return __builtin_amdgcn_wmma_f32_16x16x32_bf16(false, A.v, false, B.v,
                                                   (short)0, C, false, false);
}

// A fragment (16x32 bf16, M=lane&15): VGPR0-7 = K kb..kb+15,
// kb = 16 for lanes 16-31. 16 contiguous bf16 -> two 16B loads.
__device__ __forceinline__ void loadA(FragB& A, const unsigned short* ar) {
    A.q[0] = *(const u32x4*)(ar);
    A.q[1] = *(const u32x4*)(ar + 8);
}

// B fragment (32x16 bf16, N=lane&15): lanes 0-15 hold K 0..15, lanes 16-31 hold
// K 16..31, 2 consecutive K per VGPR. Raw path: wp points at row (k0+kbB), col c.
__device__ __forceinline__ void loadB_raw(FragB& B, const float* wp, int ld) {
#pragma unroll
    for (int j = 0; j < 8; ++j) {
        B.ui[j] = packbf_fast(wp[(size_t)(2 * j) * ld], wp[(size_t)(2 * j + 1) * ld]);
    }
}

// Tiled path: bt points at this lane's 16 contiguous bf16 of the current k-tile.
__device__ __forceinline__ void loadB_tiled(FragB& B, const unsigned short* bt) {
    B.q[0] = *(const u32x4*)(bt);
    B.q[1] = *(const u32x4*)(bt + 8);
}

// ------------------------------------------------------------- fp32 -> bf16 (x)
__global__ __launch_bounds__(256) void k_cvt(const float* __restrict__ x,
                                             unsigned short* __restrict__ xb, int n) {
    int i = blockIdx.x * 256 + threadIdx.x;
    if (i < n) xb[i] = f2bf(x[i]);
}

__global__ __launch_bounds__(256) void k_zero(unsigned short* __restrict__ p, int n) {
    int i = blockIdx.x * 256 + threadIdx.x;
    if (i < n) p[i] = 0;
}

// ---------------------------------------------- weight fp32 -> bf16 fragment tiles
// W: [batch][Kdim][Ndim] row-major fp32.  WT: [batch][nTiles][kTiles][512] bf16,
// tile (nt,kt): lane l holds W[kt*32 + (l&16?16:0) + j][nt*16 + (l&15)], j=0..15.
__global__ __launch_bounds__(256) void k_tilew(const float* __restrict__ W,
                                               unsigned short* __restrict__ WT,
                                               int Kdim, int Ndim,
                                               int nTiles, int kTiles) {
    const int wv = threadIdx.x >> 5;
    const int lane = threadIdx.x & 31;
    const int t = blockIdx.x * 8 + wv;
    if (t >= nTiles * kTiles) return;
    const int nt = t / kTiles;
    const int kt = t - nt * kTiles;
    const int kb = (lane & 16) ? 16 : 0;
    const float* src = W + (size_t)blockIdx.y * Kdim * Ndim +
                       (size_t)(kt * 32 + kb) * Ndim + nt * 16 + (lane & 15);
    unsigned short* dst = WT + ((size_t)blockIdx.y * nTiles * kTiles + t) * 512 + lane * 16;
    FragB f;
#pragma unroll
    for (int j = 0; j < 8; ++j) {
        f.ui[j] = packbf_rne(src[(size_t)(2 * j) * Ndim], src[(size_t)(2 * j + 1) * Ndim]);
    }
    *(u32x4*)dst = f.q[0];
    *(u32x4*)(dst + 8) = f.q[1];
}

// ---------------------------------------------------------------- gating
__global__ __launch_bounds__(64) void k_gate(const float* __restrict__ x,
                                             const float* __restrict__ gw,
                                             int* __restrict__ flat_i,
                                             float* __restrict__ flat_w) {
    const int t = blockIdx.x;
    const int e = threadIdx.x;
    const float* xr = x + (size_t)t * HDIM;
    const float* wr = gw + (size_t)e * HDIM;
    float acc = 0.f;
    for (int h = 0; h < HDIM; h += 4) {
        acc += xr[h] * wr[h] + xr[h + 1] * wr[h + 1] +
               xr[h + 2] * wr[h + 2] + xr[h + 3] * wr[h + 3];
    }
    __shared__ float lg[NEXP];
    lg[e] = acc;
    __syncthreads();
    if (e == 0) {
        float mx = lg[0];
        for (int i = 1; i < NEXP; ++i) mx = lg[i] > mx ? lg[i] : mx;
        float sum = 0.f;
        for (int i = 0; i < NEXP; ++i) sum += __expf(lg[i] - mx);
        bool used[NEXP];
        for (int i = 0; i < NEXP; ++i) used[i] = false;
        for (int kk = 0; kk < TOPK; ++kk) {
            int bi = 0;
            float bv = -1e30f;
            for (int i = 0; i < NEXP; ++i) {
                if (!used[i] && lg[i] > bv) { bv = lg[i]; bi = i; }
            }
            used[bi] = true;
            flat_i[t * TOPK + kk] = bi;
            flat_w[t * TOPK + kk] = __expf(bv - mx) / sum;
        }
    }
}

// -------------------------------------------- stable per-expert rank + dispatch maps
__global__ __launch_bounds__(64) void k_rank(const int* __restrict__ fi,
                                             float* __restrict__ fw,
                                             int* __restrict__ cnt,
                                             int* __restrict__ etok,
                                             int* __restrict__ eslot) {
    const int e = threadIdx.x;
    int c = 0;
    for (int s = 0; s < TKTOT; ++s) {
        if (fi[s] == e) {
            if (c < CAP) {
                etok[e * CAP + c]  = s / TOPK;
                eslot[e * CAP + c] = s;
            } else {
                fw[s] = 0.f;
            }
            ++c;
        }
    }
    cnt[e] = c < CAP ? c : CAP;
}

// ---------------------------------------------------------------- expert gate+up
template<bool TILED>
__global__ __launch_bounds__(256) void k_gateup(const unsigned short* __restrict__ xb,
                                                const unsigned short* __restrict__ zrow,
                                                const float* __restrict__ wg,
                                                const unsigned short* __restrict__ wgT,
                                                const float* __restrict__ wu,
                                                const unsigned short* __restrict__ wuT,
                                                const int* __restrict__ cnt,
                                                const int* __restrict__ etok,
                                                unsigned short* __restrict__ sbuf) {
    const int e = blockIdx.x;
    const int count = cnt[e];
    const int m0 = blockIdx.z * 32;
    if (m0 >= count) return;
    const int wv = threadIdx.x >> 5;
    const int lane = threadIdx.x & 31;
    const int lo = lane & 15;
    const int hi = (lane >> 4) & 1;
    const int nt = blockIdx.y * 8 + wv;   // 16-col stripe index
    const int c = nt * 16 + lo;
    const int kbA = hi * 16;

    const int r0 = m0 + lo;
    const int r1 = m0 + 16 + lo;
    // out-of-range rows read a zero-filled dummy row -> branch-free inner loop
    const unsigned short* a0 =
        (r0 < count) ? xb + (size_t)etok[e * CAP + r0] * HDIM + kbA : zrow + kbA;
    const unsigned short* a1 =
        (r1 < count) ? xb + (size_t)etok[e * CAP + r1] * HDIM + kbA : zrow + kbA;

    v8f g0 = {0.f, 0.f, 0.f, 0.f, 0.f, 0.f, 0.f, 0.f};
    v8f g1 = g0, u0 = g0, u1 = g0;

    if constexpr (TILED) {
        const int nTiles = IDIM / 16, kTiles = HDIM / 32;
        const unsigned short* bg =
            wgT + (((size_t)e * nTiles + nt) * kTiles) * 512 + lane * 16;
        const unsigned short* bu =
            wuT + (((size_t)e * nTiles + nt) * kTiles) * 512 + lane * 16;
        // software-pipelined ping-pong: set 'a' holds k0, set 'b' holds k0+32
        FragB A0a, A1a, Bga, Bua, A0b, A1b, Bgb, Bub;
        loadA(A0a, a0);
        loadA(A1a, a1);
        loadB_tiled(Bga, bg);
        loadB_tiled(Bua, bu);
        for (int k0 = 0; k0 < HDIM - 64; k0 += 64) {
            loadA(A0b, a0 + k0 + 32);
            loadA(A1b, a1 + k0 + 32);
            loadB_tiled(Bgb, bg + 512);
            loadB_tiled(Bub, bu + 512);
            __builtin_prefetch(bg + 4096, 0, 1);
            __builtin_prefetch(bu + 4096, 0, 1);
            g0 = wmma_bf16(A0a, Bga, g0);
            g1 = wmma_bf16(A1a, Bga, g1);
            u0 = wmma_bf16(A0a, Bua, u0);
            u1 = wmma_bf16(A1a, Bua, u1);
            loadA(A0a, a0 + k0 + 64);
            loadA(A1a, a1 + k0 + 64);
            loadB_tiled(Bga, bg + 1024);
            loadB_tiled(Bua, bu + 1024);
            bg += 1024; bu += 1024;
            g0 = wmma_bf16(A0b, Bgb, g0);
            g1 = wmma_bf16(A1b, Bgb, g1);
            u0 = wmma_bf16(A0b, Bub, u0);
            u1 = wmma_bf16(A1b, Bub, u1);
        }
        loadA(A0b, a0 + HDIM - 32);
        loadA(A1b, a1 + HDIM - 32);
        loadB_tiled(Bgb, bg + 512);
        loadB_tiled(Bub, bu + 512);
        g0 = wmma_bf16(A0a, Bga, g0);
        g1 = wmma_bf16(A1a, Bga, g1);
        u0 = wmma_bf16(A0a, Bua, u0);
        u1 = wmma_bf16(A1a, Bua, u1);
        g0 = wmma_bf16(A0b, Bgb, g0);
        g1 = wmma_bf16(A1b, Bgb, g1);
        u0 = wmma_bf16(A0b, Bub, u0);
        u1 = wmma_bf16(A1b, Bub, u1);
    } else {
        const int kbB = hi * 16;
        const float* wgp = wg + (size_t)e * HDIM * IDIM + c;
        const float* wup = wu + (size_t)e * HDIM * IDIM + c;
        for (int k0 = 0; k0 < HDIM; k0 += 32) {
            FragB A0, A1, Bg, Bu;
            loadA(A0, a0 + k0);
            loadA(A1, a1 + k0);
            loadB_raw(Bg, wgp + (size_t)(k0 + kbB) * IDIM, IDIM);
            loadB_raw(Bu, wup + (size_t)(k0 + kbB) * IDIM, IDIM);
            g0 = wmma_bf16(A0, Bg, g0);
            g1 = wmma_bf16(A1, Bg, g1);
            u0 = wmma_bf16(A0, Bu, u0);
            u1 = wmma_bf16(A1, Bu, u1);
        }
    }

    unsigned short* sp = sbuf + (size_t)e * CAP * IDIM + c;
    const int rb0 = m0 + hi * 8;
#pragma unroll
    for (int i = 0; i < 8; ++i) {
        sp[(size_t)(rb0 + i) * IDIM]      = f2bf(silu_f(g0[i]) * u0[i]);
        sp[(size_t)(rb0 + 16 + i) * IDIM] = f2bf(silu_f(g1[i]) * u1[i]);
    }
}

// ---------------------------------------------------------------- expert down
template<bool TILED>
__global__ __launch_bounds__(256) void k_down(const unsigned short* __restrict__ sbuf,
                                              const float* __restrict__ wd,
                                              const unsigned short* __restrict__ wdT,
                                              const int* __restrict__ cnt,
                                              const int* __restrict__ eslot,
                                              float* __restrict__ ysl) {
    const int e = blockIdx.x;
    const int count = cnt[e];
    const int m0 = blockIdx.z * 32;
    if (m0 >= count) return;
    const int wv = threadIdx.x >> 5;
    const int lane = threadIdx.x & 31;
    const int lo = lane & 15;
    const int hi = (lane >> 4) & 1;
    const int nt = blockIdx.y * 8 + wv;
    const int c = nt * 16 + lo;
    const int kbA = hi * 16;

    const unsigned short* a0 = sbuf + (size_t)(e * CAP + m0 + lo) * IDIM + kbA;
    const unsigned short* a1 = a0 + (size_t)16 * IDIM;

    v8f acc0 = {0.f, 0.f, 0.f, 0.f, 0.f, 0.f, 0.f, 0.f};
    v8f acc1 = acc0;

    if constexpr (TILED) {
        const int nTiles = HDIM / 16, kTiles = IDIM / 32;
        const unsigned short* bp =
            wdT + (((size_t)e * nTiles + nt) * kTiles) * 512 + lane * 16;
        FragB A0a, A1a, Ba, A0b, A1b, Bb;
        loadA(A0a, a0);
        loadA(A1a, a1);
        loadB_tiled(Ba, bp);
        for (int k0 = 0; k0 < IDIM - 64; k0 += 64) {
            loadA(A0b, a0 + k0 + 32);
            loadA(A1b, a1 + k0 + 32);
            loadB_tiled(Bb, bp + 512);
            __builtin_prefetch(bp + 4096, 0, 1);
            acc0 = wmma_bf16(A0a, Ba, acc0);
            acc1 = wmma_bf16(A1a, Ba, acc1);
            loadA(A0a, a0 + k0 + 64);
            loadA(A1a, a1 + k0 + 64);
            loadB_tiled(Ba, bp + 1024);
            bp += 1024;
            acc0 = wmma_bf16(A0b, Bb, acc0);
            acc1 = wmma_bf16(A1b, Bb, acc1);
        }
        loadA(A0b, a0 + IDIM - 32);
        loadA(A1b, a1 + IDIM - 32);
        loadB_tiled(Bb, bp + 512);
        acc0 = wmma_bf16(A0a, Ba, acc0);
        acc1 = wmma_bf16(A1a, Ba, acc1);
        acc0 = wmma_bf16(A0b, Bb, acc0);
        acc1 = wmma_bf16(A1b, Bb, acc1);
    } else {
        const int kbB = hi * 16;
        const float* wp = wd + (size_t)e * IDIM * HDIM + c;
        for (int k0 = 0; k0 < IDIM; k0 += 32) {
            FragB A0, A1, B;
            loadA(A0, a0 + k0);
            loadA(A1, a1 + k0);
            loadB_raw(B, wp + (size_t)(k0 + kbB) * HDIM, HDIM);
            acc0 = wmma_bf16(A0, B, acc0);
            acc1 = wmma_bf16(A1, B, acc1);
        }
    }

    const int rb0 = m0 + hi * 8;
#pragma unroll
    for (int i = 0; i < 8; ++i) {
        int r = rb0 + i;
        if (r < count) ysl[(size_t)eslot[e * CAP + r] * HDIM + c] = acc0[i];
        r = rb0 + 16 + i;
        if (r < count) ysl[(size_t)eslot[e * CAP + r] * HDIM + c] = acc1[i];
    }
}

// ---------------------------------------------------------------- shared gate+up
template<bool TILED>
__global__ __launch_bounds__(256) void k_shgateup(const unsigned short* __restrict__ xb,
                                                  const float* __restrict__ swg,
                                                  const unsigned short* __restrict__ sgT,
                                                  const float* __restrict__ swu,
                                                  const unsigned short* __restrict__ suT,
                                                  unsigned short* __restrict__ ssb) {
    const int m0 = blockIdx.y * 32;
    const int wv = threadIdx.x >> 5;
    const int lane = threadIdx.x & 31;
    const int lo = lane & 15;
    const int hi = (lane >> 4) & 1;
    const int nt = blockIdx.x * 8 + wv;
    const int c = nt * 16 + lo;
    const int kbA = hi * 16;

    const unsigned short* a0 = xb + (size_t)(m0 + lo) * HDIM + kbA;
    const unsigned short* a1 = a0 + (size_t)16 * HDIM;

    v8f g0 = {0.f, 0.f, 0.f, 0.f, 0.f, 0.f, 0.f, 0.f};
    v8f g1 = g0, u0 = g0, u1 = g0;

    if constexpr (TILED) {
        const int kTiles = HDIM / 32;
        const unsigned short* bg = sgT + ((size_t)nt * kTiles) * 512 + lane * 16;
        const unsigned short* bu = suT + ((size_t)nt * kTiles) * 512 + lane * 16;
        FragB A0a, A1a, Bga, Bua, A0b, A1b, Bgb, Bub;
        loadA(A0a, a0);
        loadA(A1a, a1);
        loadB_tiled(Bga, bg);
        loadB_tiled(Bua, bu);
        for (int k0 = 0; k0 < HDIM - 64; k0 += 64) {
            loadA(A0b, a0 + k0 + 32);
            loadA(A1b, a1 + k0 + 32);
            loadB_tiled(Bgb, bg + 512);
            loadB_tiled(Bub, bu + 512);
            __builtin_prefetch(bg + 4096, 0, 1);
            __builtin_prefetch(bu + 4096, 0, 1);
            g0 = wmma_bf16(A0a, Bga, g0);
            g1 = wmma_bf16(A1a, Bga, g1);
            u0 = wmma_bf16(A0a, Bua, u0);
            u1 = wmma_bf16(A1a, Bua, u1);
            loadA(A0a, a0 + k0 + 64);
            loadA(A1a, a1 + k0 + 64);
            loadB_tiled(Bga, bg + 1024);
            loadB_tiled(Bua, bu + 1024);
            bg += 1024; bu += 1024;
            g0 = wmma_bf16(A0b, Bgb, g0);
            g1 = wmma_bf16(A1b, Bgb, g1);
            u0 = wmma_bf16(A0b, Bub, u0);
            u1 = wmma_bf16(A1b, Bub, u1);
        }
        loadA(A0b, a0 + HDIM - 32);
        loadA(A1b, a1 + HDIM - 32);
        loadB_tiled(Bgb, bg + 512);
        loadB_tiled(Bub, bu + 512);
        g0 = wmma_bf16(A0a, Bga, g0);
        g1 = wmma_bf16(A1a, Bga, g1);
        u0 = wmma_bf16(A0a, Bua, u0);
        u1 = wmma_bf16(A1a, Bua, u1);
        g0 = wmma_bf16(A0b, Bgb, g0);
        g1 = wmma_bf16(A1b, Bgb, g1);
        u0 = wmma_bf16(A0b, Bub, u0);
        u1 = wmma_bf16(A1b, Bub, u1);
    } else {
        const int kbB = hi * 16;
        const float* gp = swg + c;
        const float* up = swu + c;
        for (int k0 = 0; k0 < HDIM; k0 += 32) {
            FragB A0, A1, Bg, Bu;
            loadA(A0, a0 + k0);
            loadA(A1, a1 + k0);
            loadB_raw(Bg, gp + (size_t)(k0 + kbB) * ISH, ISH);
            loadB_raw(Bu, up + (size_t)(k0 + kbB) * ISH, ISH);
            g0 = wmma_bf16(A0, Bg, g0);
            g1 = wmma_bf16(A1, Bg, g1);
            u0 = wmma_bf16(A0, Bu, u0);
            u1 = wmma_bf16(A1, Bu, u1);
        }
    }

    unsigned short* sp = ssb + c;
    const int rb0 = m0 + hi * 8;
#pragma unroll
    for (int i = 0; i < 8; ++i) {
        sp[(size_t)(rb0 + i) * ISH]      = f2bf(silu_f(g0[i]) * u0[i]);
        sp[(size_t)(rb0 + 16 + i) * ISH] = f2bf(silu_f(g1[i]) * u1[i]);
    }
}

// --------------------------------------- shared down GEMM + top-k combine epilogue
template<bool TILED>
__global__ __launch_bounds__(256) void k_final(const unsigned short* __restrict__ ssb,
                                               const float* __restrict__ swd,
                                               const unsigned short* __restrict__ sdT,
                                               const float* __restrict__ ysl,
                                               const float* __restrict__ fw,
                                               float* __restrict__ out) {
    const int m0 = blockIdx.y * 32;
    const int wv = threadIdx.x >> 5;
    const int lane = threadIdx.x & 31;
    const int lo = lane & 15;
    const int hi = (lane >> 4) & 1;
    const int nt = blockIdx.x * 8 + wv;
    const int c = nt * 16 + lo;
    const int kbA = hi * 16;

    const unsigned short* a0 = ssb + (size_t)(m0 + lo) * ISH + kbA;
    const unsigned short* a1 = a0 + (size_t)16 * ISH;

    v8f acc0 = {0.f, 0.f, 0.f, 0.f, 0.f, 0.f, 0.f, 0.f};
    v8f acc1 = acc0;

    if constexpr (TILED) {
        const int kTiles = ISH / 32;
        const unsigned short* bp = sdT + ((size_t)nt * kTiles) * 512 + lane * 16;
        FragB A0a, A1a, Ba, A0b, A1b, Bb;
        loadA(A0a, a0);
        loadA(A1a, a1);
        loadB_tiled(Ba, bp);
        for (int k0 = 0; k0 < ISH - 64; k0 += 64) {
            loadA(A0b, a0 + k0 + 32);
            loadA(A1b, a1 + k0 + 32);
            loadB_tiled(Bb, bp + 512);
            __builtin_prefetch(bp + 4096, 0, 1);
            acc0 = wmma_bf16(A0a, Ba, acc0);
            acc1 = wmma_bf16(A1a, Ba, acc1);
            loadA(A0a, a0 + k0 + 64);
            loadA(A1a, a1 + k0 + 64);
            loadB_tiled(Ba, bp + 1024);
            bp += 1024;
            acc0 = wmma_bf16(A0b, Bb, acc0);
            acc1 = wmma_bf16(A1b, Bb, acc1);
        }
        loadA(A0b, a0 + ISH - 32);
        loadA(A1b, a1 + ISH - 32);
        loadB_tiled(Bb, bp + 512);
        acc0 = wmma_bf16(A0a, Ba, acc0);
        acc1 = wmma_bf16(A1a, Ba, acc1);
        acc0 = wmma_bf16(A0b, Bb, acc0);
        acc1 = wmma_bf16(A1b, Bb, acc1);
    } else {
        const int kbB = hi * 16;
        const float* wp = swd + c;
        for (int k0 = 0; k0 < ISH; k0 += 32) {
            FragB A0, A1, B;
            loadA(A0, a0 + k0);
            loadA(A1, a1 + k0);
            loadB_raw(B, wp + (size_t)(k0 + kbB) * HDIM, HDIM);
            acc0 = wmma_bf16(A0, B, acc0);
            acc1 = wmma_bf16(A1, B, acc1);
        }
    }

    const int rb0 = m0 + hi * 8;
#pragma unroll
    for (int i = 0; i < 8; ++i) {
        int t = rb0 + i;
        float v = acc0[i];
#pragma unroll
        for (int kk = 0; kk < TOPK; ++kk) {
            const int s = t * TOPK + kk;
            v += fw[s] * ysl[(size_t)s * HDIM + c];
        }
        out[(size_t)t * HDIM + c] = v;

        t = rb0 + 16 + i;
        v = acc1[i];
#pragma unroll
        for (int kk = 0; kk < TOPK; ++kk) {
            const int s = t * TOPK + kk;
            v += fw[s] * ysl[(size_t)s * HDIM + c];
        }
        out[(size_t)t * HDIM + c] = v;
    }
}

// ----------------------------------------------------------------------- host
extern "C" void kernel_launch(void* const* d_in, const int* in_sizes, int n_in,
                              void* d_out, int out_size, void* d_ws, size_t ws_size,
                              hipStream_t stream) {
    (void)in_sizes; (void)n_in; (void)out_size;
    const float* x   = (const float*)d_in[0];
    const float* gw  = (const float*)d_in[1];
    const float* wg  = (const float*)d_in[2];
    const float* wu  = (const float*)d_in[3];
    const float* wd  = (const float*)d_in[4];
    const float* swg = (const float*)d_in[5];
    const float* swu = (const float*)d_in[6];
    const float* swd = (const float*)d_in[7];
    float* out = (float*)d_out;

    char* ws = (char*)d_ws;
    size_t off = 0;
    auto alloc = [&](size_t bytes) -> void* {
        off = (off + 255) & ~(size_t)255;
        void* p = ws + off;
        off += bytes;
        return p;
    };

    unsigned short* xb   = (unsigned short*)alloc((size_t)T_TOK * HDIM * 2);
    unsigned short* zrow = (unsigned short*)alloc((size_t)HDIM * 2);
    int*            fi   = (int*)alloc((size_t)TKTOT * 4);
    float*          fwt  = (float*)alloc((size_t)TKTOT * 4);
    int*            cnt  = (int*)alloc((size_t)NEXP * 4);
    int*            etok = (int*)alloc((size_t)NEXP * CAP * 4);
    int*            eslt = (int*)alloc((size_t)NEXP * CAP * 4);
    unsigned short* sbuf = (unsigned short*)alloc((size_t)NEXP * CAP * IDIM * 2);
    unsigned short* ssb  = (unsigned short*)alloc((size_t)T_TOK * ISH * 2);
    float*          ysl  = (float*)alloc((size_t)TKTOT * HDIM * 4);

    // tiled bf16 weight buffers (used only if workspace is large enough)
    unsigned short* wgT = (unsigned short*)alloc((size_t)NEXP * HDIM * IDIM * 2);
    unsigned short* wuT = (unsigned short*)alloc((size_t)NEXP * HDIM * IDIM * 2);
    unsigned short* wdT = (unsigned short*)alloc((size_t)NEXP * IDIM * HDIM * 2);
    unsigned short* sgT = (unsigned short*)alloc((size_t)HDIM * ISH * 2);
    unsigned short* suT = (unsigned short*)alloc((size_t)HDIM * ISH * 2);
    unsigned short* sdT = (unsigned short*)alloc((size_t)ISH * HDIM * 2);
    const bool tiled = (off <= ws_size);

    k_cvt<<<(T_TOK * HDIM + 255) / 256, 256, 0, stream>>>(x, xb, T_TOK * HDIM);
    k_zero<<<(HDIM + 255) / 256, 256, 0, stream>>>(zrow, HDIM);
    k_gate<<<T_TOK, 64, 0, stream>>>(x, gw, fi, fwt);
    k_rank<<<1, 64, 0, stream>>>(fi, fwt, cnt, etok, eslt);

    if (tiled) {
        // nTiles = N/16, kTiles = K/32; blocks = nTiles*kTiles/8 (all exact)
        k_tilew<<<dim3((IDIM / 16) * (HDIM / 32) / 8, NEXP), 256, 0, stream>>>(
            wg, wgT, HDIM, IDIM, IDIM / 16, HDIM / 32);
        k_tilew<<<dim3((IDIM / 16) * (HDIM / 32) / 8, NEXP), 256, 0, stream>>>(
            wu, wuT, HDIM, IDIM, IDIM / 16, HDIM / 32);
        k_tilew<<<dim3((HDIM / 16) * (IDIM / 32) / 8, NEXP), 256, 0, stream>>>(
            wd, wdT, IDIM, HDIM, HDIM / 16, IDIM / 32);
        k_tilew<<<dim3((ISH / 16) * (HDIM / 32) / 8, 1), 256, 0, stream>>>(
            swg, sgT, HDIM, ISH, ISH / 16, HDIM / 32);
        k_tilew<<<dim3((ISH / 16) * (HDIM / 32) / 8, 1), 256, 0, stream>>>(
            swu, suT, HDIM, ISH, ISH / 16, HDIM / 32);
        k_tilew<<<dim3((HDIM / 16) * (ISH / 32) / 8, 1), 256, 0, stream>>>(
            swd, sdT, ISH, HDIM, HDIM / 16, ISH / 32);

        k_gateup<true><<<dim3(NEXP, 11, 8), 256, 0, stream>>>(xb, zrow, wg, wgT, wu, wuT,
                                                              cnt, etok, sbuf);
        k_down<true><<<dim3(NEXP, 16, 8), 256, 0, stream>>>(sbuf, wd, wdT, cnt, eslt, ysl);
        k_shgateup<true><<<dim3(22, 32), 256, 0, stream>>>(xb, swg, sgT, swu, suT, ssb);
        k_final<true><<<dim3(16, 32), 256, 0, stream>>>(ssb, swd, sdT, ysl, fwt, out);
    } else {
        k_gateup<false><<<dim3(NEXP, 11, 8), 256, 0, stream>>>(xb, zrow, wg, wgT, wu, wuT,
                                                               cnt, etok, sbuf);
        k_down<false><<<dim3(NEXP, 16, 8), 256, 0, stream>>>(sbuf, wd, wdT, cnt, eslt, ysl);
        k_shgateup<false><<<dim3(22, 32), 256, 0, stream>>>(xb, swg, sgT, swu, suT, ssb);
        k_final<false><<<dim3(16, 32), 256, 0, stream>>>(ssb, swd, sdT, ysl, fwt, out);
    }
}
